// VarianceCoding_16741782520300
// MI455X (gfx1250) — compile-verified, matrix operand order checked by reading, not run on hardware
//
#include <hip/hip_runtime.h>
#include <cstdint>

// ---------------------------------------------------------------------------
// VarianceCoding on MI455X (gfx1250): 9 fused proximal-RMSProp steps.
// bf16 WMMA GEMMs with f32 accumulate; async global->LDS double buffering.
// ---------------------------------------------------------------------------

typedef __attribute__((ext_vector_type(16))) __bf16 v16bf;
typedef __attribute__((ext_vector_type(8)))  float  v8f;
typedef __attribute__((ext_vector_type(4)))  int    v4i;

typedef __attribute__((address_space(1))) v4i as1_v4i;
typedef __attribute__((address_space(3))) v4i as3_v4i;

#define VC_B    4096
#define VC_IN   2048
#define VC_OUT  4096
#define VC_LR   0.009f

#if defined(__AMDGCN__) && \
    __has_builtin(__builtin_amdgcn_global_load_async_to_lds_b128) && \
    __has_builtin(__builtin_amdgcn_s_wait_asynccnt)
#define VC_ASYNC 1
#else
#define VC_ASYNC 0
#endif

#if VC_ASYNC
__device__ __forceinline__ void vc_async_b128(const __bf16* g, __bf16* l) {
  // global_load_async_to_lds_b128 : HBM -> LDS without touching VGPRs
  __builtin_amdgcn_global_load_async_to_lds_b128(
      (as1_v4i*)(uintptr_t)g,
      (as3_v4i*)(uint32_t)(uintptr_t)l,
      0, 0);
}
#endif

// ---------------------------------------------------------------------------
// One-time: W fp32 [OUT][IN] -> Wb bf16 [OUT][IN] and Wt bf16 [IN][OUT]
// ---------------------------------------------------------------------------
__global__ void vc_convert_w(const float* __restrict__ W,
                             __bf16* __restrict__ Wb, __bf16* __restrict__ Wt,
                             int outd, int ind) {
  int i = blockIdx.x * blockDim.x + threadIdx.x;
  if (i >= outd * ind) return;
  int o = i / ind, c = i - o * ind;
  float w = W[i];
  Wb[i] = (__bf16)w;
  Wt[(size_t)c * outd + o] = (__bf16)w;
}

// ---------------------------------------------------------------------------
// One-time: zero x / acc1 / acc2 / x_bf16 (workspace arrives poisoned)
// ---------------------------------------------------------------------------
__global__ void vc_zero_state(float* __restrict__ x, float* __restrict__ a1,
                              float* __restrict__ a2, __bf16* __restrict__ xb,
                              int n) {
  int i = blockIdx.x * blockDim.x + threadIdx.x;
  if (i >= n) return;
  x[i] = 0.0f; a1[i] = 0.0f; a2[i] = 0.0f; xb[i] = (__bf16)0.0f;
}

// ---------------------------------------------------------------------------
// Tiled bf16 WMMA GEMM, C = A[M,K] * B[K,N] (B passed transposed: Bt[N][K]),
// f32 accumulate, fused epilogue:
//   EP==0 : gpre = -0.05 * exp(-u) * sqrt(in^2 + 1e-6)      (bf16 out)
//   EP==1 : RMSProp + prox update of x/acc1/acc2 (+ bf16 x mirror)
//   EP==2 : out  =  0.05 * (1 + exp(-u)) * sqrt(in^2 + 1e-6) (f32 out)
// Block: 256 threads (8 wave32), 256x128 tile, wave tile 64x64, K-tile = 32.
// Double-buffered LDS fed by async global->LDS DMA (ASYNCcnt).
// ---------------------------------------------------------------------------
template<int EP>
__global__ __launch_bounds__(256)
void vc_gemm_bf16_wmma(const __bf16* __restrict__ A,   // [M][K]
                       const __bf16* __restrict__ Bt,  // [N][K]
                       int M, int N, int K,
                       const float* __restrict__ inp,  // EP0/EP2: inputs [M][N]
                       __bf16* __restrict__ gpre,      // EP0 out
                       float*  __restrict__ xv,        // EP1 in/out
                       float*  __restrict__ a1v,       // EP1 in/out
                       float*  __restrict__ a2v,       // EP1 in/out
                       __bf16* __restrict__ xbf,       // EP1 out
                       float*  __restrict__ outv) {    // EP2 out
  constexpr int LDA = 40;                       // padded LDS stride (bf16)
  __shared__ __bf16 As[2][256 * LDA];           // A tiles [m][k], double buffer
  __shared__ __bf16 Bs[2][128 * LDA];           // B tiles [n][k], double buffer

  const int tid   = threadIdx.x;
  const int lane  = tid & 31;
  const int wid   = tid >> 5;          // 0..7
  const int warpM = wid & 3;           // 4 waves along M  (rows: warpM*64)
  const int warpN = wid >> 2;          // 2 waves along N  (cols: warpN*64)
  const int lrow  = lane & 15;
  const int half  = lane >> 4;         // 0: K 0..7/16..23   1: K 8..15/24..31
  const int kBase = half * 8;

  const int bm = blockIdx.y * 256;
  const int bn = blockIdx.x * 128;

  // warm L2 for the fused epilogue operands while the GEMM runs
  {
    const size_t pidx = (size_t)(bm + tid) * N + bn;
    if constexpr (EP == 1) {
      __builtin_prefetch(xv  + pidx, 0, 1);
      __builtin_prefetch(a1v + pidx, 0, 1);
      __builtin_prefetch(a2v + pidx, 0, 1);
    } else {
      __builtin_prefetch(inp + pidx, 0, 1);
    }
  }

  const v8f vzero = {0.f, 0.f, 0.f, 0.f, 0.f, 0.f, 0.f, 0.f};
  v8f acc[4][4];
#pragma unroll
  for (int i = 0; i < 4; ++i)
#pragma unroll
    for (int j = 0; j < 4; ++j) acc[i][j] = vzero;

  // Stage one K-tile (A: 256x32, B: 128x32) into LDS buffer `buf`.
  auto stage = [&](int k0, int buf) {
#if VC_ASYNC
#pragma unroll
    for (int it = 0; it < 4; ++it) {            // A: 1024 b128 chunks / 256 thr
      const int c = tid + it * 256;
      const int row = c >> 2, col8 = (c & 3) << 3;
      vc_async_b128(A + (size_t)(bm + row) * K + k0 + col8,
                    &As[buf][row * LDA + col8]);
    }
#pragma unroll
    for (int it = 0; it < 2; ++it) {            // B: 512 b128 chunks / 256 thr
      const int c = tid + it * 256;
      const int row = c >> 2, col8 = (c & 3) << 3;
      vc_async_b128(Bt + (size_t)(bn + row) * K + k0 + col8,
                    &Bs[buf][row * LDA + col8]);
    }
#else
    uint4 ra[4], rb[2];
#pragma unroll
    for (int it = 0; it < 4; ++it) {
      const int c = tid + it * 256;
      ra[it] = *reinterpret_cast<const uint4*>(
          A + (size_t)(bm + (c >> 2)) * K + k0 + ((c & 3) << 3));
    }
#pragma unroll
    for (int it = 0; it < 2; ++it) {
      const int c = tid + it * 256;
      rb[it] = *reinterpret_cast<const uint4*>(
          Bt + (size_t)(bn + (c >> 2)) * K + k0 + ((c & 3) << 3));
    }
#pragma unroll
    for (int it = 0; it < 4; ++it) {
      const int c = tid + it * 256;
      *reinterpret_cast<uint4*>(&As[buf][(c >> 2) * LDA + ((c & 3) << 3)]) = ra[it];
    }
#pragma unroll
    for (int it = 0; it < 2; ++it) {
      const int c = tid + it * 256;
      *reinterpret_cast<uint4*>(&Bs[buf][(c >> 2) * LDA + ((c & 3) << 3)]) = rb[it];
    }
#endif
  };

  stage(0, 0);

  int buf = 0;
  for (int k0 = 0; k0 < K; k0 += 32, buf ^= 1) {
#if VC_ASYNC
    __builtin_amdgcn_s_wait_asynccnt(0);        // my DMA into `buf` landed
#endif
    __syncthreads();                             // everyone's tile is in LDS
    if (k0 + 32 < K) stage(k0 + 32, buf ^ 1);    // overlap next DMA with WMMA

    // ---- fragments per documented 16-bit A-matrix lane layout ----
    v16bf afrag[4], bfrag[4];
#pragma unroll
    for (int i = 0; i < 4; ++i) {
      const int r = warpM * 64 + i * 16 + lrow;
      uint4* p = reinterpret_cast<uint4*>(&afrag[i]);
      p[0] = *reinterpret_cast<const uint4*>(&As[buf][r * LDA + kBase]);
      p[1] = *reinterpret_cast<const uint4*>(&As[buf][r * LDA + kBase + 16]);
    }
#pragma unroll
    for (int j = 0; j < 4; ++j) {
      const int c = warpN * 64 + j * 16 + lrow;
      uint4* p = reinterpret_cast<uint4*>(&bfrag[j]);
      p[0] = *reinterpret_cast<const uint4*>(&Bs[buf][c * LDA + kBase]);
      p[1] = *reinterpret_cast<const uint4*>(&Bs[buf][c * LDA + kBase + 16]);
    }

    // ---- 16 x v_wmma_f32_16x16x32_bf16 ----
#pragma unroll
    for (int i = 0; i < 4; ++i)
#pragma unroll
      for (int j = 0; j < 4; ++j)
        acc[i][j] = __builtin_amdgcn_wmma_f32_16x16x32_bf16(
            false, afrag[i], false, bfrag[j], (short)0, acc[i][j],
            false, false);
  }

  // ---- fused epilogue (C/D layout: VGPR e -> row half*8+e, col lrow) ----
#pragma unroll
  for (int i = 0; i < 4; ++i) {
#pragma unroll
    for (int j = 0; j < 4; ++j) {
      const int rowBase = bm + warpM * 64 + i * 16 + half * 8;
      const int col     = bn + warpN * 64 + j * 16 + lrow;
#pragma unroll
      for (int e = 0; e < 8; ++e) {
        const size_t idx = (size_t)(rowBase + e) * N + col;
        const float u = acc[i][j][e];
        if constexpr (EP == 0) {
          const float t  = inp[idx];
          const float aa = __builtin_sqrtf(t * t + 1e-6f);
          gpre[idx] = (__bf16)(-0.05f * __expf(-u) * aa);
        } else if constexpr (EP == 1) {
          const float xo = xv[idx];
          // + d(l1_inov)/dx + d(own_norm)/dx
          const float g    = u + 0.02f * xo + 0.1f * xo * rsqrtf(xo * xo + 1e-6f);
          const float a1   = 0.9f * a1v[idx] + 0.1f * g * g;
          const float inv  = rsqrtf(a1 + 1e-8f);
          const float stp  = VC_LR * g * inv;
          const float a2   = 0.9f * a2v[idx] - stp;
          const float cand = xo - stp;
          const float thr  = 0.1f * VC_LR * inv;
          const float xn   = fmaxf(cand - thr, 0.0f) + fminf(cand + thr, 0.0f)
                           + 0.9f * a2;
          xv[idx] = xn; a1v[idx] = a1; a2v[idx] = a2; xbf[idx] = (__bf16)xn;
        } else {
          const float t  = inp[idx];
          const float aa = __builtin_sqrtf(t * t + 1e-6f);
          outv[idx] = 0.05f * (1.0f + __expf(-u)) * aa;
        }
      }
    }
  }
}

// ---------------------------------------------------------------------------
extern "C" void kernel_launch(void* const* d_in, const int* in_sizes, int n_in,
                              void* d_out, int out_size, void* d_ws, size_t ws_size,
                              hipStream_t stream) {
  (void)in_sizes; (void)n_in; (void)out_size; (void)ws_size;
  const float* inputs = (const float*)d_in[0];   // [B, IN]
  const float* W      = (const float*)d_in[1];   // [OUT, IN]
  float* out = (float*)d_out;                    // [B, IN]

  char* ws = (char*)d_ws;
  __bf16* Wb = (__bf16*)(ws);                          // 16 MiB  [OUT][IN]
  __bf16* Wt = (__bf16*)(ws + ((size_t)16  << 20));    // 16 MiB  [IN][OUT]
  float*  x  = (float*) (ws + ((size_t)32  << 20));    // 64 MiB  [B][OUT]
  float*  a1 = (float*) (ws + ((size_t)96  << 20));    // 64 MiB
  float*  a2 = (float*) (ws + ((size_t)160 << 20));    // 64 MiB
  __bf16* xb = (__bf16*)(ws + ((size_t)224 << 20));    // 32 MiB  [B][OUT]
  __bf16* gp = (__bf16*)(ws + ((size_t)256 << 20));    // 16 MiB  [B][IN]

  const int nW = VC_OUT * VC_IN;
  vc_convert_w<<<(nW + 255) / 256, 256, 0, stream>>>(W, Wb, Wt, VC_OUT, VC_IN);
  const int nX = VC_B * VC_OUT;
  vc_zero_state<<<(nX + 255) / 256, 256, 0, stream>>>(x, a1, a2, xb, nX);

  const dim3 blk(256);
  const dim3 g1(VC_IN  / 128, VC_B / 256);   // (16, 16)
  const dim3 g2(VC_OUT / 128, VC_B / 256);   // (32, 16)

  for (int s = 0; s < 9; ++s) {  // N_STEPS - 1 updates
    // u = x @ W  -> g_pre   (A = x_bf16 [B][OUT], Bt = Wt [IN][OUT])
    vc_gemm_bf16_wmma<0><<<g1, blk, 0, stream>>>(
        xb, Wt, VC_B, VC_IN, VC_OUT, inputs, gp,
        nullptr, nullptr, nullptr, nullptr, nullptr);
    // g = g_pre @ W^T -> RMSProp/prox update (A = gp [B][IN], Bt = Wb [OUT][IN])
    vc_gemm_bf16_wmma<1><<<g2, blk, 0, stream>>>(
        gp, Wb, VC_B, VC_OUT, VC_IN, nullptr, nullptr,
        x, a1, a2, xb, nullptr);
  }
  // final l1_norm = 0.1 * act(x@W) * a
  vc_gemm_bf16_wmma<2><<<g1, blk, 0, stream>>>(
      xb, Wt, VC_B, VC_IN, VC_OUT, inputs, nullptr,
      nullptr, nullptr, nullptr, nullptr, out);
}
